// MultiHeadMaskedAttention_89378269430189
// MI455X (gfx1250) — matrix operand order repaired and checked by reading, not compiled, not run on hardware
//
#include <hip/hip_runtime.h>
#include <math.h>

// ---------------- problem constants ----------------
#define D_IN  768
#define NH    12
#define HD    64
#define BB    4
#define SS    2048

typedef __bf16 bf16;
typedef __attribute__((ext_vector_type(16))) __bf16 v16bf;
typedef __attribute__((ext_vector_type(8)))  float  v8f;

union AF { v16bf v; unsigned u[8]; bf16 h[16]; };
union CF { v8f  v; float f[8]; };

__device__ __forceinline__ v8f wmma_bf16(v16bf a, v16bf b, v8f c) {
  // v_wmma_f32_16x16x32_bf16: D = A(16x32) * B(32x16) + C(16x16 f32)
  return __builtin_amdgcn_wmma_f32_16x16x32_bf16(
      /*neg_a=*/false, a, /*neg_b=*/false, b,
      /*c_mod=*/(short)0, c, /*reuse_a=*/false, /*reuse_b=*/false);
}

// A-matrix 16x32 bf16 fragment (contiguous k, same order as B): lane row M = lane&15, half=lane>>4.
// elem i -> K = half*16 + i.
__device__ __forceinline__ void load_frag_a(AF& f, const bf16* rowp, int lh) {
#pragma unroll
  for (int v = 0; v < 8; ++v) {
    int k = lh * 16 + v * 2;
    f.u[v] = *(const unsigned*)(rowp + k);
  }
}

// B-matrix 32x16 bf16 layout: lane col N = lane&15; K = half*16 + 2v + e.
// rowp points at Bt[n][kbase] (K contiguous per row -> dword/b128 loads).
__device__ __forceinline__ void load_frag_b(AF& f, const bf16* rowp, int lh) {
#pragma unroll
  for (int v = 0; v < 8; ++v)
    f.u[v] = *(const unsigned*)(rowp + lh * 16 + v * 2);
}

// ---------------- conversion kernels ----------------
__global__ void f32_to_bf16_kernel(const float* __restrict__ in,
                                   bf16* __restrict__ out, int n) {
  int i = blockIdx.x * 256 + threadIdx.x;
  if (i < n) out[i] = (bf16)in[i];
}

// [G][K][N] f32 -> [G][N][K] bf16
__global__ void conv_transpose_kernel(const float* __restrict__ in,
                                      bf16* __restrict__ out,
                                      int G, int K, int N) {
  int i = blockIdx.x * 256 + threadIdx.x;
  int total = G * K * N;
  if (i >= total) return;
  int g = i / (K * N);
  int rem = i % (K * N);
  int kk = rem / N;
  int nn = rem % N;
  out[((size_t)g * N + nn) * K + kk] = (bf16)in[i];
}

// ---------- QKV projection, 4x2 register-blocked (64x32 per wave) ----------
__global__ __launch_bounds__(64)
void qkv_gemm_kernel(const bf16* __restrict__ xb,
                     const bf16* __restrict__ WqT, const bf16* __restrict__ WkT,
                     const bf16* __restrict__ WvT,
                     const float* __restrict__ bq, const float* __restrict__ bk,
                     const float* __restrict__ bv,
                     bf16* __restrict__ q, bf16* __restrict__ k,
                     bf16* __restrict__ v) {
  int wave = threadIdx.x >> 5;           // 0..1 -> 32-col slab of N=64
  int lane = threadIdx.x & 31;
  int lh = lane >> 4, l15 = lane & 15;

  int mblk = blockIdx.x;                 // 0..S/64-1
  int z = blockIdx.y;                    // 0..3*B*NH-1
  int matsel = z / (BB * NH);
  int bh = z % (BB * NH);
  int h = bh % NH;

  const bf16* WT; const float* bias; bf16* out;
  if (matsel == 0)      { WT = WqT; bias = bq; out = q; }
  else if (matsel == 1) { WT = WkT; bias = bk; out = k; }
  else                  { WT = WvT; bias = bv; out = v; }

  int b = bh / NH;
  const bf16* A  = xb + (size_t)b * SS * D_IN;
  const bf16* Bt = WT + (size_t)h * HD * D_IN;

  const bf16* arow[4];
#pragma unroll
  for (int mt = 0; mt < 4; ++mt)
    arow[mt] = A + (size_t)(mblk * 64 + mt * 16 + l15) * D_IN;
  const bf16* brow[2];
#pragma unroll
  for (int nt = 0; nt < 2; ++nt)
    brow[nt] = Bt + (size_t)(wave * 32 + nt * 16 + l15) * D_IN;

  CF acc[4][2];
#pragma unroll
  for (int mt = 0; mt < 4; ++mt)
#pragma unroll
    for (int nt = 0; nt < 2; ++nt)
#pragma unroll
      for (int i = 0; i < 8; ++i) acc[mt][nt].f[i] = 0.f;

  for (int kb = 0; kb < D_IN; kb += 32) {
    AF bfr[2];
#pragma unroll
    for (int nt = 0; nt < 2; ++nt) load_frag_b(bfr[nt], brow[nt] + kb, lh);
#pragma unroll
    for (int mt = 0; mt < 4; ++mt) {
      AF a;
      load_frag_a(a, arow[mt] + kb, lh);
#pragma unroll
      for (int nt = 0; nt < 2; ++nt)
        acc[mt][nt].v = wmma_bf16(a.v, bfr[nt].v, acc[mt][nt].v);
    }
  }

  bf16* obase = out + (size_t)bh * SS * HD;
#pragma unroll
  for (int nt = 0; nt < 2; ++nt) {
    int col = wave * 32 + nt * 16 + l15;
    float bia = bias[h * HD + col];
#pragma unroll
    for (int mt = 0; mt < 4; ++mt)
#pragma unroll
      for (int r = 0; r < 8; ++r) {
        int row = mblk * 64 + mt * 16 + r + 8 * lh;
        obase[(size_t)row * HD + col] = (bf16)(acc[mt][nt].f[r] + bia);
      }
  }
}

// ---------- causal flash attention: 1 wave handles 32 query rows ----------
__global__ __launch_bounds__(32)
void flash_attn_kernel(const bf16* __restrict__ q, const bf16* __restrict__ k,
                       const bf16* __restrict__ v, bf16* __restrict__ ctxb) {
  __shared__ bf16 Pl[2][16][32];   // P tiles: C-layout -> A-layout transpose
  __shared__ bf16 Vl[32][64];      // V tile staged via async LDS copy

  int lane = threadIdx.x & 31;
  int lh = lane >> 4, l15 = lane & 15;
  int qt = blockIdx.x;             // 32-row query block
  int h = blockIdx.y;
  int b = blockIdx.z;
  int bh = b * NH + h;

  const bf16* qp = q + (size_t)bh * SS * HD;
  const bf16* kp = k + (size_t)bh * SS * HD;
  const bf16* vp = v + (size_t)bh * SS * HD;

  int qbase = qt * 32;

  AF aq[2][2];                     // [m-tile][head-dim 0..31 / 32..63]
#pragma unroll
  for (int mt = 0; mt < 2; ++mt) {
    const bf16* qrow = qp + (size_t)(qbase + mt * 16 + l15) * HD;
    load_frag_a(aq[mt][0], qrow, lh);
    load_frag_a(aq[mt][1], qrow + 32, lh);
  }

  CF acc[2][4];
#pragma unroll
  for (int mt = 0; mt < 2; ++mt)
#pragma unroll
    for (int t = 0; t < 4; ++t)
#pragma unroll
      for (int i = 0; i < 8; ++i) acc[mt][t].f[i] = 0.f;

  float m[2][8], l[2][8];
#pragma unroll
  for (int mt = 0; mt < 2; ++mt)
#pragma unroll
    for (int r = 0; r < 8; ++r) { m[mt][r] = -__builtin_inff(); l[mt][r] = 0.f; }

  const float scale = 0.125f;      // 1/sqrt(64)
  int send = qbase + 32;           // exclusive causal bound; always full steps
  unsigned vlds = (unsigned)(uintptr_t)&Vl[0][0];   // LDS byte offset

  for (int sb = 0; sb < send; sb += 32) {
    if (sb + 32 < send) {
      __builtin_prefetch(kp + (size_t)(sb + 32) * HD, 0, 1);  // global_prefetch_b8
      __builtin_prefetch(vp + (size_t)(sb + 32) * HD, 0, 1);
    }

    // ---- async-stage V tile (32 keys x 64 dims, bf16) into LDS ----
    // Prior-iteration LDS reads must drain before the async engine rewrites Vl.
    asm volatile("s_wait_dscnt 0x0" ::: "memory");
#pragma unroll
    for (int p = 0; p < 8; ++p) {
      int idx = p * 32 + lane;               // 0..255 octet index
      int srow = idx >> 3, oct = idx & 7;
      const bf16* gaddr = vp + (size_t)(sb + srow) * HD + oct * 8;
      unsigned laddr = vlds + (unsigned)(srow * HD + oct * 8) * 2;
      asm volatile("global_load_async_to_lds_b128 %0, %1, off"
                   :: "v"(laddr), "v"(gaddr) : "memory");
    }

    // ---- scores S = Q * K^T : K fragments shared across both q m-tiles ----
    AF bk0[2], bk1[2];
#pragma unroll
    for (int c = 0; c < 2; ++c) {
      const bf16* krow = kp + (size_t)(sb + c * 16 + l15) * HD;
      load_frag_b(bk0[c], krow, lh);
      load_frag_b(bk1[c], krow + 32, lh);
    }

#pragma unroll
    for (int mt = 0; mt < 2; ++mt) {
      CF sc[2];
#pragma unroll
      for (int c = 0; c < 2; ++c) {
#pragma unroll
        for (int i = 0; i < 8; ++i) sc[c].f[i] = 0.f;
        sc[c].v = wmma_bf16(aq[mt][0].v, bk0[c].v, sc[c].v);
        sc[c].v = wmma_bf16(aq[mt][1].v, bk1[c].v, sc[c].v);
      }

      // ---- scale + causal mask ----
      float p0[8], p1[8];
#pragma unroll
      for (int r = 0; r < 8; ++r) {
        int grow = qbase + mt * 16 + r + 8 * lh;
        p0[r] = (sb + l15      <= grow) ? sc[0].f[r] * scale : -__builtin_inff();
        p1[r] = (sb + 16 + l15 <= grow) ? sc[1].f[r] * scale : -__builtin_inff();
      }

      // ---- online softmax (row stats across 16 lanes per half) ----
#pragma unroll
      for (int r = 0; r < 8; ++r) {
        float x = fmaxf(p0[r], p1[r]);
#pragma unroll
        for (int off = 8; off; off >>= 1) x = fmaxf(x, __shfl_xor(x, off, 16));
        float mn = fmaxf(m[mt][r], x);
        float alpha = __expf(m[mt][r] - mn);
        p0[r] = __expf(p0[r] - mn);
        p1[r] = __expf(p1[r] - mn);
        float s = p0[r] + p1[r];
#pragma unroll
        for (int off = 8; off; off >>= 1) s += __shfl_xor(s, off, 16);
        l[mt][r] = l[mt][r] * alpha + s;
        m[mt][r] = mn;
#pragma unroll
        for (int t = 0; t < 4; ++t) acc[mt][t].f[r] *= alpha;
      }

      // ---- P tile to LDS (C-layout -> A-layout) ----
      __syncthreads();             // single-wave WG: ordering fence only
#pragma unroll
      for (int r = 0; r < 8; ++r) {
        Pl[mt][r + 8 * lh][l15]      = (bf16)p0[r];
        Pl[mt][r + 8 * lh][16 + l15] = (bf16)p1[r];
      }
      __syncthreads();
    }

    // ---- wait for the async V tile, then ctx += P * V ----
    asm volatile("s_wait_asynccnt 0x0" ::: "memory");

    AF vf[4];
#pragma unroll
    for (int t = 0; t < 4; ++t)
#pragma unroll
      for (int i = 0; i < 16; ++i)
        vf[t].h[i] = Vl[lh * 16 + i][t * 16 + l15];   // B layout: K = lh*16 + i

#pragma unroll
    for (int mt = 0; mt < 2; ++mt) {
      AF pa;
      load_frag_a(pa, &Pl[mt][l15][0], lh);
#pragma unroll
      for (int t = 0; t < 4; ++t)
        acc[mt][t].v = wmma_bf16(pa.v, vf[t].v, acc[mt][t].v);
    }
  }

  // ---- epilogue: normalize, write ctx interleaved [b][s][h*64+n] ----
#pragma unroll
  for (int mt = 0; mt < 2; ++mt)
#pragma unroll
    for (int t = 0; t < 4; ++t)
#pragma unroll
      for (int r = 0; r < 8; ++r) {
        int row = qbase + mt * 16 + r + 8 * lh;
        float val = acc[mt][t].f[r] / l[mt][r];
        ctxb[((size_t)b * SS + row) * D_IN + h * HD + t * 16 + l15] = (bf16)val;
      }
}

// ---------- output projection, 4x2 register-blocked (64x32 per wave) --------
__global__ __launch_bounds__(128)
void out_proj_kernel(const bf16* __restrict__ ctxb, const bf16* __restrict__ WoT,
                     const float* __restrict__ bo, float* __restrict__ out) {
  int wave = threadIdx.x >> 5;           // 4 waves -> 4 N-slabs of 32
  int lane = threadIdx.x & 31;
  int lh = lane >> 4, l15 = lane & 15;

  int mblk = blockIdx.x;                 // 0..B*S/64-1
  int nslab = blockIdx.y * 4 + wave;     // 0..23

  const bf16* arow[4];
#pragma unroll
  for (int mt = 0; mt < 4; ++mt)
    arow[mt] = ctxb + (size_t)(mblk * 64 + mt * 16 + l15) * D_IN;
  const bf16* brow[2];
#pragma unroll
  for (int nt = 0; nt < 2; ++nt)
    brow[nt] = WoT + (size_t)(nslab * 32 + nt * 16 + l15) * D_IN;

  CF acc[4][2];
#pragma unroll
  for (int mt = 0; mt < 4; ++mt)
#pragma unroll
    for (int nt = 0; nt < 2; ++nt)
#pragma unroll
      for (int i = 0; i < 8; ++i) acc[mt][nt].f[i] = 0.f;

  for (int kb = 0; kb < D_IN; kb += 32) {
    AF bfr[2];
#pragma unroll
    for (int nt = 0; nt < 2; ++nt) load_frag_b(bfr[nt], brow[nt] + kb, lh);
#pragma unroll
    for (int mt = 0; mt < 4; ++mt) {
      AF a;
      load_frag_a(a, arow[mt] + kb, lh);
#pragma unroll
      for (int nt = 0; nt < 2; ++nt)
        acc[mt][nt].v = wmma_bf16(a.v, bfr[nt].v, acc[mt][nt].v);
    }
  }

#pragma unroll
  for (int nt = 0; nt < 2; ++nt) {
    int col = nslab * 32 + nt * 16 + l15;
    float bia = bo[col];
#pragma unroll
    for (int mt = 0; mt < 4; ++mt)
#pragma unroll
      for (int r = 0; r < 8; ++r) {
        int row = mblk * 64 + mt * 16 + r + 8 * lh;
        out[(size_t)row * D_IN + col] = acc[mt][nt].f[r] + bia;
      }
  }
}

// ---------------- host side ----------------
extern "C" void kernel_launch(void* const* d_in, const int* in_sizes, int n_in,
                              void* d_out, int out_size, void* d_ws, size_t ws_size,
                              hipStream_t stream) {
  const float* x  = (const float*)d_in[0];
  const float* Wq = (const float*)d_in[1];
  const float* Wk = (const float*)d_in[2];
  const float* Wv = (const float*)d_in[3];
  const float* bq = (const float*)d_in[4];
  const float* bk = (const float*)d_in[5];
  const float* bv = (const float*)d_in[6];
  const float* Wo = (const float*)d_in[7];
  const float* bo = (const float*)d_in[8];
  float* out = (float*)d_out;

  size_t off = 0;
  auto carve = [&](size_t bytes) {
    size_t o = off;
    off = (off + bytes + 255) & ~(size_t)255;
    return o;
  };
  char* ws = (char*)d_ws;
  const size_t xN   = (size_t)BB * SS * D_IN;
  const size_t wN   = (size_t)NH * D_IN * HD;
  const size_t woN  = (size_t)D_IN * D_IN;
  const size_t qkvN = (size_t)BB * NH * SS * HD;

  bf16* xb  = (bf16*)(ws + carve(xN * 2));
  bf16* WqT = (bf16*)(ws + carve(wN * 2));
  bf16* WkT = (bf16*)(ws + carve(wN * 2));
  bf16* WvT = (bf16*)(ws + carve(wN * 2));
  bf16* WoT = (bf16*)(ws + carve(woN * 2));
  bf16* qb  = (bf16*)(ws + carve(qkvN * 2));
  bf16* kb  = (bf16*)(ws + carve(qkvN * 2));
  bf16* vb  = (bf16*)(ws + carve(qkvN * 2));
  bf16* ctx = (bf16*)(ws + carve(xN * 2));
  (void)ws_size; (void)n_in; (void)in_sizes; (void)out_size;

  // 1) fp32 -> bf16 conversions / weight transposes
  f32_to_bf16_kernel<<<(int)((xN + 255) / 256), 256, 0, stream>>>(x, xb, (int)xN);
  conv_transpose_kernel<<<(int)((wN + 255) / 256), 256, 0, stream>>>(Wq, WqT, NH, D_IN, HD);
  conv_transpose_kernel<<<(int)((wN + 255) / 256), 256, 0, stream>>>(Wk, WkT, NH, D_IN, HD);
  conv_transpose_kernel<<<(int)((wN + 255) / 256), 256, 0, stream>>>(Wv, WvT, NH, D_IN, HD);
  conv_transpose_kernel<<<(int)((woN + 255) / 256), 256, 0, stream>>>(Wo, WoT, 1, D_IN, D_IN);

  // 2) QKV projection (WMMA bf16, 64x32 per wave)
  qkv_gemm_kernel<<<dim3(SS / 64, 3 * BB * NH), 64, 0, stream>>>(
      xb, WqT, WkT, WvT, bq, bk, bv, qb, kb, vb);

  // 3) causal flash attention (WMMA bf16 + async LDS staging)
  flash_attn_kernel<<<dim3(SS / 32, NH, BB), 32, 0, stream>>>(qb, kb, vb, ctx);

  // 4) output projection (WMMA bf16 -> f32 out)
  out_proj_kernel<<<dim3(BB * SS / 64, 6), 128, 0, stream>>>(ctx, WoT, bo, out);
}
